// Attention_8718783611466
// MI455X (gfx1250) — matrix operand order repaired and checked by reading, not compiled, not run on hardware
//
#include <hip/hip_runtime.h>

// Flash-attention for [2,16,2048,64] fp32 on gfx1250 (wave32, WMMA 16x16x32 f16).
// Per wave: 16 query rows, 64-key tiles. S^T = K * Q^T so softmax is lane-local;
// O^T = V^T * P^T. Double-buffered K/V tiles in LDS; xor16 via v_permlanex16.
// WMMAs issued in 4 independent chains so the matrix pipe never stalls on RAW.

typedef _Float16 v4h  __attribute__((ext_vector_type(4)));
typedef _Float16 v8h  __attribute__((ext_vector_type(8)));
typedef _Float16 v16h __attribute__((ext_vector_type(16)));
typedef float    v8f  __attribute__((ext_vector_type(8)));

#define SEQ   2048
#define HD    64
#define KT    64           // keys per tile
#define NT    (SEQ / KT)   // 32 key tiles
#define QSTR  72           // LDS row stride (halves), multiple of 8
#define KSTR  72
#define VSTR  72

// A/B fragments are two contiguous 8-half (16B) runs -> ds_load_b128 x2
static __device__ inline v16h ld2(const _Float16* p0, const _Float16* p1) {
    v8h a = *(const v8h*)p0;
    v8h b = *(const v8h*)p1;
    v16h r;
#pragma unroll
    for (int i = 0; i < 8; ++i) { r[i] = a[i]; r[i + 8] = b[i]; }
    return r;
}

static __device__ inline v8f wmma_f16(v16h a, v16h b, v8f c) {
    return __builtin_amdgcn_wmma_f32_16x16x32_f16(
        /*neg_a=*/false, a, /*neg_b=*/false, b,
        /*c_mod=*/(short)0, c, /*reuse_a=*/false, /*reuse_b=*/false);
}

// xor-16 lane exchange within wave32 as a single VALU op (v_permlanex16_b32)
static __device__ inline float xor16(float x) {
    int v = __builtin_bit_cast(int, x);
    int r = __builtin_amdgcn_permlanex16(v, v, 0x76543210, 0xfedcba98, false, false);
    return __builtin_bit_cast(float, r);
}

// Stage one 64-key K/V tile into LDS (f16, V transposed). Waves 0-3: K, waves 4-7: V.
static __device__ inline void stage_kv(const float* __restrict__ K,
                                       const float* __restrict__ V,
                                       size_t base, int kt,
                                       _Float16* __restrict__ ksb,
                                       _Float16* __restrict__ vsb, int tid) {
    const int k0 = kt * KT;
    const int kp = (kt + 2 < NT ? kt + 2 : kt) * KT;   // prefetch distance 2
    // wave-uniform split: waves 0-3 stage K, waves 4-7 stage V (scalar branch)
    if (__builtin_amdgcn_readfirstlane(tid) < 128) {
#pragma unroll
        for (int i = 0; i < 8; ++i) {
            int flat = i * 128 + tid;                  // 1024 float4 per tile
            int key = flat >> 4, dq = flat & 15;
            float4 k4 = *(const float4*)(K + base + (size_t)(k0 + key) * HD + dq * 4);
            v4h h;
            h[0] = (_Float16)k4.x; h[1] = (_Float16)k4.y;
            h[2] = (_Float16)k4.z; h[3] = (_Float16)k4.w;
            *(v4h*)&ksb[key * KSTR + dq * 4] = h;
            __builtin_prefetch(K + base + (size_t)(kp + key) * HD + dq * 4, 0, 3);
        }
    } else {
        int t2 = tid - 128;
#pragma unroll
        for (int j = 0; j < 2; ++j) {
            int b  = j * 128 + t2;                     // 256 4x4 blocks per tile
            int kb = (b & 15) * 4;                     // 4-key block
            int dq = b >> 4;                           // d quad (0..15)
            const float* vp = V + base + (size_t)(k0 + kb) * HD + dq * 4;
            float4 r0 = *(const float4*)(vp);
            float4 r1 = *(const float4*)(vp + HD);
            float4 r2 = *(const float4*)(vp + 2 * HD);
            float4 r3 = *(const float4*)(vp + 3 * HD);
            v4h c0 = {(_Float16)r0.x, (_Float16)r1.x, (_Float16)r2.x, (_Float16)r3.x};
            v4h c1 = {(_Float16)r0.y, (_Float16)r1.y, (_Float16)r2.y, (_Float16)r3.y};
            v4h c2 = {(_Float16)r0.z, (_Float16)r1.z, (_Float16)r2.z, (_Float16)r3.z};
            v4h c3 = {(_Float16)r0.w, (_Float16)r1.w, (_Float16)r2.w, (_Float16)r3.w};
            *(v4h*)&vsb[(dq * 4 + 0) * VSTR + kb] = c0;
            *(v4h*)&vsb[(dq * 4 + 1) * VSTR + kb] = c1;
            *(v4h*)&vsb[(dq * 4 + 2) * VSTR + kb] = c2;
            *(v4h*)&vsb[(dq * 4 + 3) * VSTR + kb] = c3;
            __builtin_prefetch(V + base + (size_t)(kp + kb) * HD + dq * 4, 0, 3);
        }
    }
}

__global__ __launch_bounds__(256)
void attn_fa_gfx1250(const float* __restrict__ Q, const float* __restrict__ K,
                     const float* __restrict__ V, float* __restrict__ O) {
    const int bh   = blockIdx.y;            // batch*head (0..31)
    const int qt   = blockIdx.x;            // 128-row query tile
    const int tid  = threadIdx.x;
    const int wave = tid >> 5;
    const int lane = tid & 31;
    const int hi   = lane >> 4;             // half-wave select
    const int ln   = lane & 15;
    const size_t base = (size_t)bh * SEQ * HD;
    const int q0 = qt * 128 + wave * 16;

    __shared__ __align__(16) _Float16 Qs[8][16 * QSTR];   // per-wave Q tile, f16, pre-scaled
    __shared__ __align__(16) _Float16 Ks[2][KT * KSTR];   // K tile   [key][d], double-buffered
    __shared__ __align__(16) _Float16 Vs[2][HD * VSTR];   // V tile^T [d][key], double-buffered

    // ---- stage Q tile (fold 1/sqrt(64) * log2(e) so softmax uses exp2) ----
    const float qscale = 0.125f * 1.44269504f;
#pragma unroll
    for (int i = 0; i < 8; ++i) {
        int flat = i * 32 + lane;           // 256 float4 per tile, 8 per lane
        int row = flat >> 4, dq = flat & 15;
        float4 q4 = *(const float4*)(Q + base + (size_t)(q0 + row) * HD + dq * 4);
        v4h h;
        h[0] = (_Float16)(q4.x * qscale); h[1] = (_Float16)(q4.y * qscale);
        h[2] = (_Float16)(q4.z * qscale); h[3] = (_Float16)(q4.w * qscale);
        *(v4h*)&Qs[wave][row * QSTR + dq * 4] = h;
    }
    // B fragments of Q^T (rows = d, cols = 16 queries): d runs [8h,8h+8) and [16+8h,24+8h) per lane-half
    v16h bQ0, bQ1;
    {
        const _Float16* qrow = &Qs[wave][ln * QSTR];
        const int d0 = hi * 8;
        bQ0 = ld2(qrow + d0,      qrow + d0 + 16);       // d = 0..31
        bQ1 = ld2(qrow + 32 + d0, qrow + 32 + d0 + 16);  // d = 32..63
    }

    // ---- prologue: stage tile 0 ----
    stage_kv(K, V, base, 0, Ks[0], Vs[0], tid);

    float m_run = -3.0e38f, l_run = 0.0f;
    v8f Oacc[4] = {};                       // O^T, 4 d-chunks of 16

    for (int kt = 0; kt < NT; ++kt) {
        const int cur = kt & 1;
        __syncthreads();                    // buf[cur] ready; buf[cur^1] free to overwrite
        if (kt + 1 < NT)
            stage_kv(K, V, base, kt + 1, Ks[cur ^ 1], Vs[cur ^ 1], tid);

        // ---- S^T = K_tile * Q^T : c0..c3 = keys 0..15 / 16..31 / 32..47 / 48..63 ----
        v8f c0, c1, c2, c3;
        {
            const int d0 = hi * 8;          // A runs: [d0,d0+8) and [d0+16,d0+24)
            const _Float16* kr0 = &Ks[cur][(ln +  0) * KSTR];
            const _Float16* kr1 = &Ks[cur][(ln + 16) * KSTR];
            const _Float16* kr2 = &Ks[cur][(ln + 32) * KSTR];
            const _Float16* kr3 = &Ks[cur][(ln + 48) * KSTR];
            v8f z = {};
            // phase A: d = 0..31, 4 independent WMMAs
            v16h a0 = ld2(kr0 + d0, kr0 + d0 + 16);
            v16h a1 = ld2(kr1 + d0, kr1 + d0 + 16);
            v16h a2 = ld2(kr2 + d0, kr2 + d0 + 16);
            v16h a3 = ld2(kr3 + d0, kr3 + d0 + 16);
            c0 = wmma_f16(a0, bQ0, z);
            c1 = wmma_f16(a1, bQ0, z);
            c2 = wmma_f16(a2, bQ0, z);
            c3 = wmma_f16(a3, bQ0, z);
            // phase B: d = 32..63, 4 independent accumulating WMMAs
            a0 = ld2(kr0 + 32 + d0, kr0 + 32 + d0 + 16);
            a1 = ld2(kr1 + 32 + d0, kr1 + 32 + d0 + 16);
            a2 = ld2(kr2 + 32 + d0, kr2 + 32 + d0 + 16);
            a3 = ld2(kr3 + 32 + d0, kr3 + 32 + d0 + 16);
            c0 = wmma_f16(a0, bQ1, c0);
            c1 = wmma_f16(a1, bQ1, c1);
            c2 = wmma_f16(a2, bQ1, c2);
            c3 = wmma_f16(a3, bQ1, c3);
        }

        // ---- online softmax over this 64-key tile (log2 domain) ----
        float mt = fmaxf(fmaxf(c0[0], c1[0]), fmaxf(c2[0], c3[0]));
#pragma unroll
        for (int r = 1; r < 8; ++r) {
            mt = fmaxf(mt, fmaxf(c0[r], c1[r]));
            mt = fmaxf(mt, fmaxf(c2[r], c3[r]));
        }
        mt = fmaxf(mt, xor16(mt));
        const float m_new = fmaxf(m_run, mt);
        const float corr  = __builtin_amdgcn_exp2f(m_run - m_new);
        float sum = 0.0f;
#pragma unroll
        for (int r = 0; r < 8; ++r) {
            c0[r] = __builtin_amdgcn_exp2f(c0[r] - m_new);
            c1[r] = __builtin_amdgcn_exp2f(c1[r] - m_new);
            c2[r] = __builtin_amdgcn_exp2f(c2[r] - m_new);
            c3[r] = __builtin_amdgcn_exp2f(c3[r] - m_new);
            sum += (c0[r] + c1[r]) + (c2[r] + c3[r]);
        }
        sum += xor16(sum);
        l_run = l_run * corr + sum;
        m_run = m_new;
#pragma unroll
        for (int c = 0; c < 4; ++c)
#pragma unroll
            for (int r = 0; r < 8; ++r) Oacc[c][r] *= corr;

        // ---- build B fragments of P^T: p01 = keys 0..31, p23 = keys 32..63 ----
        v16h p01, p23;
#pragma unroll
        for (int r = 0; r < 8; ++r) {
            float o0 = xor16(c0[r]);
            float o1 = xor16(c1[r]);
            float o2 = xor16(c2[r]);
            float o3 = xor16(c3[r]);
            p01[r]     = (_Float16)c0[r];
            p01[r + 8] = (_Float16)c1[r];
            p23[r]     = (_Float16)c2[r];
            p23[r + 8] = (_Float16)c3[r];
        }

        // ---- O^T += V^T * P^T : 4 independent chains x 2 key halves ----
        {
            const int kk0 = hi * 8;         // A runs: keys [kk0,kk0+8) and [kk0+16,kk0+24)
            const _Float16* vr0 = &Vs[cur][( 0 + ln) * VSTR];
            const _Float16* vr1 = &Vs[cur][(16 + ln) * VSTR];
            const _Float16* vr2 = &Vs[cur][(32 + ln) * VSTR];
            const _Float16* vr3 = &Vs[cur][(48 + ln) * VSTR];
            v16h av0 = ld2(vr0 + kk0, vr0 + kk0 + 16);
            v16h av1 = ld2(vr1 + kk0, vr1 + kk0 + 16);
            v16h av2 = ld2(vr2 + kk0, vr2 + kk0 + 16);
            v16h av3 = ld2(vr3 + kk0, vr3 + kk0 + 16);
            Oacc[0] = wmma_f16(av0, p01, Oacc[0]);
            Oacc[1] = wmma_f16(av1, p01, Oacc[1]);
            Oacc[2] = wmma_f16(av2, p01, Oacc[2]);
            Oacc[3] = wmma_f16(av3, p01, Oacc[3]);
            av0 = ld2(vr0 + 32 + kk0, vr0 + 32 + kk0 + 16);
            av1 = ld2(vr1 + 32 + kk0, vr1 + 32 + kk0 + 16);
            av2 = ld2(vr2 + 32 + kk0, vr2 + 32 + kk0 + 16);
            av3 = ld2(vr3 + 32 + kk0, vr3 + 32 + kk0 + 16);
            Oacc[0] = wmma_f16(av0, p23, Oacc[0]);
            Oacc[1] = wmma_f16(av1, p23, Oacc[1]);
            Oacc[2] = wmma_f16(av2, p23, Oacc[2]);
            Oacc[3] = wmma_f16(av3, p23, Oacc[3]);
        }
    }

    // ---- epilogue: normalize and store O ----
    const float inv_l = 1.0f / l_run;
    const size_t orow = base + (size_t)(q0 + ln) * HD;
#pragma unroll
    for (int c = 0; c < 4; ++c)
#pragma unroll
        for (int r = 0; r < 8; ++r) {
            int d = c * 16 + hi * 8 + r;
            O[orow + d] = Oacc[c][r] * inv_l;
        }
}

extern "C" void kernel_launch(void* const* d_in, const int* in_sizes, int n_in,
                              void* d_out, int out_size, void* d_ws, size_t ws_size,
                              hipStream_t stream) {
    const float* Q = (const float*)d_in[0];
    const float* K = (const float*)d_in[1];
    const float* V = (const float*)d_in[2];
    float* O = (float*)d_out;
    dim3 grid(SEQ / 128, 2 * 16);           // 16 query tiles x 32 (batch*head)
    attn_fa_gfx1250<<<grid, 256, 0, stream>>>(Q, K, V, O);
}
